// COB_Rotation_58669253263997
// MI455X (gfx1250) — matrix operand order repaired and checked by reading, not compiled, not run on hardware
//
#include <hip/hip_runtime.h>

typedef __attribute__((ext_vector_type(16))) __bf16 v16bf;
typedef __attribute__((ext_vector_type(8)))  __bf16 bf16x8;
typedef __attribute__((ext_vector_type(8)))  float  v8f;
typedef __attribute__((ext_vector_type(4)))  float  f32x4;

#define LDX 4096   // row stride of x / out in floats
#define MROWS 8192 // 4 * 2048

// Phase 1: split fp32 weights into bf16 hi/lo pair (round-to-nearest).
__global__ __launch_bounds__(256) void rot_wsplit(const float* __restrict__ w,
                                                  __bf16* __restrict__ hi,
                                                  __bf16* __restrict__ lo,
                                                  int n) {
  int i = blockIdx.x * 256 + threadIdx.x;
  if (i < n) {
    float f = w[i];
    __bf16 h = (__bf16)f;
    hi[i] = h;
    lo[i] = (__bf16)(f - (float)h);  // residual exactly representable in f32
  }
}

// Phase 2: C = X_chunk @ W^T via bf16x3 split WMMA.
// WG: 8 waves = 4 (M) x 2 (N); wave tile = 16 rows x 64 cols (4 x 16x16 accumulators).
// S is a compile-time block size so B-subtile strides fold into load immediates,
// and all offsets are 32-bit ints so loads use SADDR+VGPR32 addressing.
// All 8 B tiles (4 hi + 4 lo) are preloaded into distinct register buffers so the
// scheduler can keep every load in flight and use partial s_wait_loadcnt values
// instead of a full wait before each WMMA.
template <int S>
__global__ __launch_bounds__(256) void rot_gemm(const float* __restrict__ x,
                                                float* __restrict__ out,
                                                const __bf16* __restrict__ whi,
                                                const __bf16* __restrict__ wlo,
                                                int xoff) {
  const int tid  = threadIdx.x;
  const int lane = tid & 31;
  const int wave = tid >> 5;
  const int lr   = lane & 15;   // M (A) / N (B,C) index within tile
  const int half = lane >> 4;   // selects K-half (A,B) / M-half (C)

  const int mw = wave >> 1;     // 0..3
  const int nw = wave & 1;      // 0..1

  const int m0 = blockIdx.y * 64 + mw * 16;
  const int n0 = blockIdx.x * 128 + nw * 64;

  const v8f vzero = {0.f, 0.f, 0.f, 0.f, 0.f, 0.f, 0.f, 0.f};
  v8f acc[4];
#pragma unroll
  for (int t = 0; t < 4; ++t) acc[t] = vzero;

  // A k order used here (contiguous, same as B): lane<16 -> M=lr, K = 0..15;
  //                            lane>=16 -> M=lr, K = 16..31.
  int aoff = (m0 + lr) * LDX + xoff + half * 16;   // element offset into x

  // B k order used here (contiguous, same as A): lane<16 -> N=lr, K=0..15; lane>=16 -> N=lr, K=16..31.
  // B[k,n] = W[n,k] -> contiguous 16-bf16 run along a W row.
  int boff = (n0 + lr) * S + half * 16;           // element offset into whi/wlo

  union BU { bf16x8 h2[2]; v16bf v; };

  for (int k0 = 0; k0 < S; k0 += 32, aoff += 32, boff += 32) {
    // ---- issue all loads for this K panel up front (distinct buffers) ----
    union { f32x4 q[4]; float f[16]; } au;
    au.q[0] = *(const f32x4*)(x + aoff);
    au.q[1] = *(const f32x4*)(x + aoff + 4);
    au.q[2] = *(const f32x4*)(x + aoff + 8);
    au.q[3] = *(const f32x4*)(x + aoff + 12);

    BU bh[4], bl[4];
#pragma unroll
    for (int t = 0; t < 4; ++t) {
      const int bt = boff + t * 16 * S;  // t*16*S is a compile-time constant
      bh[t].h2[0] = *(const bf16x8*)(whi + bt);
      bh[t].h2[1] = *(const bf16x8*)(whi + bt + 8);
    }
#pragma unroll
    for (int t = 0; t < 4; ++t) {
      const int bt = boff + t * 16 * S;
      bl[t].h2[0] = *(const bf16x8*)(wlo + bt);
      bl[t].h2[1] = *(const bf16x8*)(wlo + bt + 8);
    }

    // ---- split A to bf16 hi/lo in registers (co-executes with WMMA) ----
    union { v16bf v; __bf16 e[16]; } ah, al;
#pragma unroll
    for (int e = 0; e < 16; ++e) {
      float f = au.f[e];
      __bf16 h = (__bf16)f;
      ah.e[e] = h;
      al.e[e] = (__bf16)(f - (float)h);
    }

    // ---- 12 WMMAs, ordered to match load completion order ----
#pragma unroll
    for (int t = 0; t < 4; ++t) {
      acc[t] = __builtin_amdgcn_wmma_f32_16x16x32_bf16(
          false, ah.v, false, bh[t].v, (short)0, acc[t], false, false);
      acc[t] = __builtin_amdgcn_wmma_f32_16x16x32_bf16(
          false, al.v, false, bh[t].v, (short)0, acc[t], false, false);
    }
#pragma unroll
    for (int t = 0; t < 4; ++t) {
      acc[t] = __builtin_amdgcn_wmma_f32_16x16x32_bf16(
          false, ah.v, false, bl[t].v, (short)0, acc[t], false, false);
    }
  }

  // C layout (32-bit C 16x16): VGPR r: lanes 0-15 -> M=r, lanes 16-31 -> M=8+r; N=lr.
#pragma unroll
  for (int t = 0; t < 4; ++t) {
    int ooff = (m0 + half * 8) * LDX + xoff + n0 + t * 16 + lr;
#pragma unroll
    for (int r = 0; r < 8; ++r) out[ooff + r * LDX] = acc[t][r];
  }
}

static inline void launch_gemm(int s, const float* x, float* out,
                               const __bf16* whi, const __bf16* wlo,
                               int xoff, hipStream_t stream) {
  dim3 grid(s / 128, MROWS / 64), block(256);
  switch (s) {
    case 128:  rot_gemm<128> <<<grid, block, 0, stream>>>(x, out, whi, wlo, xoff); break;
    case 256:  rot_gemm<256> <<<grid, block, 0, stream>>>(x, out, whi, wlo, xoff); break;
    case 384:  rot_gemm<384> <<<grid, block, 0, stream>>>(x, out, whi, wlo, xoff); break;
    case 512:  rot_gemm<512> <<<grid, block, 0, stream>>>(x, out, whi, wlo, xoff); break;
    case 640:  rot_gemm<640> <<<grid, block, 0, stream>>>(x, out, whi, wlo, xoff); break;
    case 768:  rot_gemm<768> <<<grid, block, 0, stream>>>(x, out, whi, wlo, xoff); break;
    case 1024: rot_gemm<1024><<<grid, block, 0, stream>>>(x, out, whi, wlo, xoff); break;
  }
}

extern "C" void kernel_launch(void* const* d_in, const int* in_sizes, int n_in,
                              void* d_out, int out_size, void* d_ws, size_t ws_size,
                              hipStream_t stream) {
  const int SIZES[8] = {512, 1024, 256, 768, 384, 640, 128, 384};

  const float* x = (const float*)d_in[0];
  float* out = (float*)d_out;

  size_t wtot = 0;
  for (int i = 0; i < 8; ++i) wtot += (size_t)SIZES[i] * SIZES[i];  // 2,686,976

  // Workspace layout: [ whi : wtot bf16 ][ wlo : wtot bf16 ]  (~10.75 MB total)
  __bf16* whi = (__bf16*)d_ws;
  __bf16* wlo = whi + wtot;

  // Phase 1: split all weights (idempotent, deterministic, runs every call).
  size_t woff = 0;
  for (int i = 0; i < 8; ++i) {
    int n = SIZES[i] * SIZES[i];
    rot_wsplit<<<dim3((n + 255) / 256), dim3(256), 0, stream>>>(
        (const float*)d_in[1 + i], whi + woff, wlo + woff, n);
    woff += n;
  }

  // Phase 2: one GEMM launch per diagonal block.
  woff = 0;
  int xoff = 0;
  for (int i = 0; i < 8; ++i) {
    int s = SIZES[i];
    launch_gemm(s, x, out, whi + woff, wlo + woff, xoff, stream);
    woff += (size_t)s * s;
    xoff += s;
  }
}
